// AttentionPool_6305011991000
// MI455X (gfx1250) — matrix operand order repaired and checked by reading, not compiled, not run on hardware
//
#include <hip/hip_runtime.h>
#include <hip/hip_bf16.h>

#define N_ATOMS 131072
#define N_MOL   4096
#define FEAT    512

typedef __attribute__((ext_vector_type(16))) _Float16 v16h;
typedef __attribute__((ext_vector_type(8)))  _Float16 v8h;
typedef __attribute__((ext_vector_type(4)))  _Float16 v4h;
typedef __attribute__((ext_vector_type(2)))  _Float16 v2h;
typedef __attribute__((ext_vector_type(8)))  float    v8f;

// ---- monotonic float<->uint key for atomicMax on possibly-negative floats ----
static __device__ __forceinline__ unsigned fkey(float f) {
    unsigned b = __float_as_uint(f);
    return (b & 0x80000000u) ? ~b : (b | 0x80000000u);
}
static __device__ __forceinline__ float funkey(unsigned k) {
    return (k & 0x80000000u) ? __uint_as_float(k & 0x7FFFFFFFu)
                             : __uint_as_float(~k);
}

// shifted softplus: softplus(x) - log(2), numerically stable
static __device__ __forceinline__ float sspf(float v) {
    float sp = fmaxf(v, 0.0f) + log1pf(expf(-fabsf(v)));
    return sp - 0.69314718055994531f;
}

// ---------------------------------------------------------------------------
// K0: convert W (512x512 f32) -> f16 workspace; init segmax/denom/starts
// ---------------------------------------------------------------------------
__global__ __launch_bounds__(256)
void k_prep(const float* __restrict__ W, _Float16* __restrict__ W16,
            unsigned* __restrict__ segmax, float* __restrict__ denom,
            int* __restrict__ starts) {
    int i = blockIdx.x * 256 + threadIdx.x;          // 0 .. 262143
    W16[i] = (_Float16)W[i];
    if (i < N_MOL) {
        segmax[i] = 0u;                // below key of any finite float
        denom[i]  = 0.0f;
        starts[i] = N_ATOMS;           // default for trailing empty segments
    }
    if (i == 0) starts[N_MOL] = N_ATOMS;
}

// ---------------------------------------------------------------------------
// K1: wf = feats @ W^T  (f16 WMMA, f32 accum), fused attention-score logits.
// Block = 256 thr (8 waves), tile = 32 atom rows x 512 cols; wave w owns
// columns [w*64, w*64+64) as 4 WMMA col-tiles for BOTH 16-row tiles, so each
// B fragment feeds 2 WMMAs. K-loop over 512 in steps of 32.
// ---------------------------------------------------------------------------
__global__ __launch_bounds__(256)
void k_gemm_scores(const float* __restrict__ feats,
                   const _Float16* __restrict__ W16,
                   const float* __restrict__ att_w,
                   _Float16* __restrict__ wf16,
                   float* __restrict__ scores) {
    __shared__ _Float16 Atile[32 * 520];   // padded stride: bank-conflict-free
    __shared__ float score_part[32];

    const int tid  = threadIdx.x;
    const int wave = tid >> 5;
    const int lane = tid & 31;
    const int g    = lane >> 4;            // lane group (0/1)
    const int ml   = lane & 15;
    const int row0 = blockIdx.x * 32;

    if (tid < 32) score_part[tid] = 0.0f;

    // ---- phase 1: cooperative load of 32x512 feats tile, f32 -> f16 in LDS
    #pragma unroll
    for (int j = 0; j < 16; ++j) {
        int idx = tid + j * 256;           // float4 index in [0, 4096)
        int r   = idx >> 7;                // 128 float4 per row
        int c4  = idx & 127;
        const float4 f =
            ((const float4*)(feats + (size_t)(row0 + r) * FEAT))[c4];
        v4h h = { (_Float16)f.x, (_Float16)f.y, (_Float16)f.z, (_Float16)f.w };
        *(v4h*)&Atile[r * 520 + c4 * 4] = h;
    }
    __syncthreads();

    // ---- phase 2: WMMA K-loop
    v8f acc[2][4];
    #pragma unroll
    for (int rt = 0; rt < 2; ++rt)
        #pragma unroll
        for (int t = 0; t < 4; ++t) acc[rt][t] = (v8f){};

    const int ncol0 = wave * 64 + ml;      // this lane's B column per tile t*16
    const _Float16* wr[4];
    #pragma unroll
    for (int t = 0; t < 4; ++t) wr[t] = W16 + (size_t)(ncol0 + 16 * t) * FEAT;

    for (int kk = 0; kk < FEAT; kk += 32) {
        // A fragments (16x32 each): lane holds row ml (+16), K = kk+8g..+7 / +16
        v8h a0lo = *(const v8h*)&Atile[ml * 520 + kk + 8 * g];
        v8h a0hi = *(const v8h*)&Atile[ml * 520 + kk + 16 + 8 * g];
        v8h a1lo = *(const v8h*)&Atile[(16 + ml) * 520 + kk + 8 * g];
        v8h a1hi = *(const v8h*)&Atile[(16 + ml) * 520 + kk + 16 + 8 * g];
        v16h a0 = __builtin_shufflevector(a0lo, a0hi,
                    0,1,2,3,4,5,6,7,8,9,10,11,12,13,14,15);
        v16h a1 = __builtin_shufflevector(a1lo, a1hi,
                    0,1,2,3,4,5,6,7,8,9,10,11,12,13,14,15);
        const int kb = kk + 8 * g;
        #pragma unroll
        for (int t = 0; t < 4; ++t) {
            // B fragment (32x16): lane holds col ncol0+t*16, K = kk+8g..+7 / +16
            v8h blo = *(const v8h*)(wr[t] + kb);
            v8h bhi = *(const v8h*)(wr[t] + kb + 16);
            v16h b = __builtin_shufflevector(blo, bhi,
                        0,1,2,3,4,5,6,7,8,9,10,11,12,13,14,15);
            acc[0][t] = __builtin_amdgcn_wmma_f32_16x16x32_f16(
                false, a0, false, b, (short)0, acc[0][t], false, false);
            acc[1][t] = __builtin_amdgcn_wmma_f32_16x16x32_f16(
                false, a1, false, b, (short)0, acc[1][t], false, false);
        }
    }
    __syncthreads();   // all waves done reading Atile

    // ---- phase 3: score partials + restage wf tile into LDS (reuse Atile)
    float aw[4];
    #pragma unroll
    for (int t = 0; t < 4; ++t) aw[t] = att_w[ncol0 + 16 * t];

    #pragma unroll
    for (int rt = 0; rt < 2; ++rt) {
        #pragma unroll
        for (int r = 0; r < 8; ++r) {
            const int m = rt * 16 + 8 * g + r;   // C/D layout: elem r <-> M=r+8g
            float sp = 0.0f;
            #pragma unroll
            for (int t = 0; t < 4; ++t) {
                float v = acc[rt][t][r];
                sp += v * aw[t];
                Atile[m * 520 + ncol0 + 16 * t] = (_Float16)v;
            }
            atomicAdd(&score_part[m], sp);       // ds_add_f32
        }
    }
    __syncthreads();

    if (tid < 32) {
        float s = score_part[tid];
        scores[row0 + tid] = (s > 0.0f) ? s : 0.01f * s;   // leaky_relu
    }
    // coalesced f16 store of the 32x512 wf tile
    #pragma unroll
    for (int j = 0; j < 8; ++j) {
        int c  = tid + j * 256;            // 16-byte chunk id in [0, 2048)
        int r  = c >> 5;                   // 32 chunks per row
        int cc = (c & 31) * 8;             // half column
        v8h v = *(const v8h*)&Atile[r * 520 + cc];
        *(v8h*)(wf16 + (size_t)(row0 + r) * FEAT + cc) = v;
    }
}

// ---------------------------------------------------------------------------
// K2a: segment max (via ordered-uint atomicMax) + segment start offsets
// ---------------------------------------------------------------------------
__global__ __launch_bounds__(256)
void k_seg_max(const int* __restrict__ seg, const float* __restrict__ scores,
               unsigned* __restrict__ segmax, int* __restrict__ starts) {
    int i = blockIdx.x * 256 + threadIdx.x;
    int s = seg[i];
    int sprev = (i == 0) ? -1 : seg[i - 1];
    for (int m = sprev + 1; m <= s; ++m) starts[m] = i;  // boundary fill
    atomicMax(&segmax[s], fkey(scores[i]));
}

// ---------------------------------------------------------------------------
// K2b: ex = exp(score - segmax), denom accumulation
// ---------------------------------------------------------------------------
__global__ __launch_bounds__(256)
void k_seg_exp(const int* __restrict__ seg, const float* __restrict__ scores,
               const unsigned* __restrict__ segmax, float* __restrict__ ex,
               float* __restrict__ denom) {
    int i = blockIdx.x * 256 + threadIdx.x;
    int m = seg[i];
    float e = expf(scores[i] - funkey(segmax[m]));
    ex[i] = e;
    atomicAdd(&denom[m], e);
}

// ---------------------------------------------------------------------------
// K3: mol_fp[m,:] = sum_i in segment m of (ex[i]/denom[m]) * wf[i,:]
// One block per molecule; thread t owns columns 2t, 2t+1.
// ---------------------------------------------------------------------------
__global__ __launch_bounds__(256)
void k_pool(const int* __restrict__ starts, const float* __restrict__ ex,
            const float* __restrict__ denom, const _Float16* __restrict__ wf16,
            float* __restrict__ molfp) {
    const int m = blockIdx.x;
    const int s = starts[m], e = starts[m + 1];
    const float den = denom[m];
    const float inv = (e > s && den != 0.0f) ? 1.0f / den : 0.0f;
    const int c = threadIdx.x * 2;
    float ax = 0.0f, ay = 0.0f;
    for (int i = s; i < e; ++i) {
        float wgt = ex[i] * inv;
        v2h hv = *(const v2h*)(wf16 + (size_t)i * FEAT + c);
        ax += wgt * (float)hv[0];
        ay += wgt * (float)hv[1];
    }
    molfp[(size_t)m * FEAT + c]     = ax;
    molfp[(size_t)m * FEAT + c + 1] = ay;
}

// ---------------------------------------------------------------------------
// K4: MLP 512 -> 170 (ssp) -> 56 (ssp) -> 1 ; one block per molecule
// ---------------------------------------------------------------------------
__global__ __launch_bounds__(256)
void k_mlp(const float* __restrict__ molfp,
           const float* __restrict__ W1, const float* __restrict__ b1,
           const float* __restrict__ W2, const float* __restrict__ b2,
           const float* __restrict__ W3, const float* __restrict__ b3,
           float* __restrict__ out) {
    __shared__ __align__(16) float x[512];
    __shared__ float h1[170];
    __shared__ float h2[56];
    const int m = blockIdx.x, t = threadIdx.x;
    x[t]       = molfp[(size_t)m * FEAT + t];
    x[t + 256] = molfp[(size_t)m * FEAT + t + 256];
    __syncthreads();
    if (t < 170) {
        const float4* wr = (const float4*)(W1 + (size_t)t * 512);
        const float4* xv = (const float4*)x;
        float acc = b1[t];
        #pragma unroll 4
        for (int k = 0; k < 128; ++k) {
            float4 w = wr[k], v = xv[k];
            acc += w.x * v.x + w.y * v.y + w.z * v.z + w.w * v.w;
        }
        h1[t] = sspf(acc);
    }
    __syncthreads();
    if (t < 56) {
        const float* wr = W2 + (size_t)t * 170;
        float acc = b2[t];
        for (int k = 0; k < 170; ++k) acc += wr[k] * h1[k];
        h2[t] = sspf(acc);
    }
    __syncthreads();
    if (t == 0) {
        float acc = b3[0];
        for (int k = 0; k < 56; ++k) acc += W3[k] * h2[k];
        out[m] = acc;
    }
}

// ---------------------------------------------------------------------------
extern "C" void kernel_launch(void* const* d_in, const int* in_sizes, int n_in,
                              void* d_out, int out_size, void* d_ws, size_t ws_size,
                              hipStream_t stream) {
    (void)in_sizes; (void)n_in; (void)out_size; (void)ws_size;
    const float* feats = (const float*)d_in[0];
    const int*   seg   = (const int*)  d_in[1];
    const float* W     = (const float*)d_in[2];
    const float* att_w = (const float*)d_in[3];
    const float* W1    = (const float*)d_in[4];
    const float* b1    = (const float*)d_in[5];
    const float* W2    = (const float*)d_in[6];
    const float* b2    = (const float*)d_in[7];
    const float* W3    = (const float*)d_in[8];
    const float* b3    = (const float*)d_in[9];

    float* out   = (float*)d_out;              // [N_MOL]
    float* molfp = (float*)d_out + N_MOL;      // [N_MOL * FEAT]

    char* ws = (char*)d_ws;
    _Float16* wf16   = (_Float16*)ws;  ws += (size_t)N_ATOMS * FEAT * 2;
    _Float16* W16    = (_Float16*)ws;  ws += (size_t)FEAT * FEAT * 2;
    float*    scores = (float*)ws;     ws += (size_t)N_ATOMS * 4;
    float*    ex     = (float*)ws;     ws += (size_t)N_ATOMS * 4;
    unsigned* segmax = (unsigned*)ws;  ws += (size_t)N_MOL * 4;
    float*    denom  = (float*)ws;     ws += (size_t)N_MOL * 4;
    int*      starts = (int*)ws;       ws += (size_t)(N_MOL + 1) * 4;

    k_prep<<<(FEAT * FEAT) / 256, 256, 0, stream>>>(W, W16, segmax, denom, starts);
    k_gemm_scores<<<N_ATOMS / 32, 256, 0, stream>>>(feats, W16, att_w, wf16, scores);
    k_seg_max<<<N_ATOMS / 256, 256, 0, stream>>>(seg, scores, segmax, starts);
    k_seg_exp<<<N_ATOMS / 256, 256, 0, stream>>>(seg, scores, segmax, ex, denom);
    k_pool<<<N_MOL, 256, 0, stream>>>(starts, ex, denom, wf16, molfp);
    k_mlp<<<N_MOL, 256, 0, stream>>>(molfp, W1, b1, W2, b2, W3, b3, out);
}
